// GCViTLayer_86904368267311
// MI455X (gfx1250) — matrix operand order repaired and checked by reading, not compiled, not run on hardware
//
#include <hip/hip_runtime.h>
#include <hip/hip_bf16.h>
#include <math.h>
#include <stdint.h>

// ---------------------------------------------------------------------------
// GCViT block forward for gfx1250 (CDNA5, wave32, WMMA + TDM).
// Shapes hardcoded from setup_inputs(): B=64,H=8,W=128,C=256,nh=8,hd=32,
// wh=8, ww=16, N=128, B_=512, tokens=65536, dff=1024.
// ---------------------------------------------------------------------------

typedef _Float16 half_t;
typedef __attribute__((ext_vector_type(8)))  _Float16 v8h;
typedef __attribute__((ext_vector_type(16))) _Float16 v16h;
typedef __attribute__((ext_vector_type(8)))  float    v8f;
typedef unsigned int v4u __attribute__((ext_vector_type(4)));
typedef int          v4i __attribute__((ext_vector_type(4)));
typedef int          v8i __attribute__((ext_vector_type(8)));

#define TOKENS   65536
#define CDIM     256
#define DFF      1024
#define NHEAD    8
#define HDIM     32
#define NWIN     128      // tokens per window
#define BWIN     512      // number of windows (B_)
#define BIAS_LD  63       // 2*MAX_WW-1

#if defined(__has_builtin)
#if __has_builtin(__builtin_amdgcn_tensor_load_to_lds) && \
    __has_builtin(__builtin_amdgcn_s_wait_tensorcnt)
#define HAVE_TDM 1
#endif
#endif

__device__ __forceinline__ v8f wmma_f16(v16h a, v16h b, v8f c) {
    // D = A(16x32 f16) x B(32x16 f16) + C(16x16 f32)
    return __builtin_amdgcn_wmma_f32_16x16x32_f16(
        /*neg_a=*/false, a, /*neg_b=*/false, b,
        /*c_mod=*/(short)0, c, /*reuse_a=*/false, /*reuse_b=*/false);
}

// A-matrix fragment (16x32 f16, row-major source, ld in elements, 16B aligned
// rows). Lane l: row = l&15; K chunks {8*half..} and {16+8*half..}, half=l>>4.
__device__ __forceinline__ v16h load_a16(const half_t* base, int ld) {
    const int lane = (int)(threadIdx.x & 31u);
    const int row  = lane & 15;
    const int hf   = lane >> 4;
    const v8h* p0 = (const v8h*)(base + row * ld + 8 * hf);
    const v8h* p1 = (const v8h*)(base + row * ld + 16 + 8 * hf);
    v8h lo = *p0, hi = *p1;
    return __builtin_shufflevector(lo, hi, 0,1,2,3,4,5,6,7,8,9,10,11,12,13,14,15);
}

// B-matrix fragment (32x16 f16). Source stored as (Ncols rows x K cols)
// row-major (i.e. a weight matrix W[out][in], keys k[n][d], or v^T[d][n]).
// Lane l: col = l&15; K chunks {8*half..} and {16+8*half..}, half=l>>4.
__device__ __forceinline__ v16h load_b16(const half_t* base, int ld) {
    const int lane = (int)(threadIdx.x & 31u);
    const int col  = lane & 15;
    const int hf   = lane >> 4;
    const v8h* p0 = (const v8h*)(base + col * ld + 8 * hf);
    const v8h* p1 = (const v8h*)(base + col * ld + 16 + 8 * hf);
    v8h lo = *p0, hi = *p1;
    return __builtin_shufflevector(lo, hi, 0,1,2,3,4,5,6,7,8,9,10,11,12,13,14,15);
}

// One wave computes a 32x32 output tile as 2x2 WMMA tiles over the full K.
__device__ __forceinline__ void wave_gemm32x32(const half_t* Ab, int lda,
                                               const half_t* Bb, int ldb,
                                               int K, v8f acc[2][2]) {
    for (int kc = 0; kc < K; kc += 32) {
        v16h a0 = load_a16(Ab + kc, lda);
        v16h a1 = load_a16(Ab + (size_t)16 * lda + kc, lda);
        v16h b0 = load_b16(Bb + kc, ldb);
        v16h b1 = load_b16(Bb + (size_t)16 * ldb + kc, ldb);
        acc[0][0] = wmma_f16(a0, b0, acc[0][0]);
        acc[0][1] = wmma_f16(a0, b1, acc[0][1]);
        acc[1][0] = wmma_f16(a1, b0, acc[1][0]);
        acc[1][1] = wmma_f16(a1, b1, acc[1][1]);
    }
}

static __device__ const v8f VZERO8 = {0.f,0.f,0.f,0.f,0.f,0.f,0.f,0.f};

#ifdef HAVE_TDM
// Issue a 1-D TDM copy of `nbytes` contiguous bytes (multiple of 4, tracked
// with TENSORcnt) from global memory to LDS byte offset `lds_off`.
// D# layout per CDNA5 ISA ch.8: group0 {count/type/addr}, group1 {sizes};
// 6-arg builtin form: (g0, g1, g2, g3, g4/VADDR4 = zeros, cpol).
__device__ __forceinline__ void tdm_load_1d(uint64_t gaddr, unsigned lds_off,
                                            unsigned nbytes) {
    const unsigned elems = nbytes >> 2;          // 4-byte elements
    v4u g0; v8i g1; v4i g2, g3; v8i g4;
    g0[0] = 1u;                                  // count=1, user descriptor
    g0[1] = lds_off;                             // lds_addr (bytes)
    g0[2] = (unsigned)(gaddr & 0xFFFFFFFFull);   // global_addr[31:0]
    g0[3] = (unsigned)((gaddr >> 32) & 0x01FFFFFFull) | (2u << 30); // type=2
    g1[0] = (int)(2u << 16);                     // data_size=4B, no mask/pad
    g1[1] = (int)((elems & 0xFFFFu) << 16);      // tensor_dim0[15:0]
    g1[2] = (int)(((elems >> 16) & 0xFFFFu) | (1u << 16)); // dim0 hi | dim1=1
    g1[3] = (int)((elems & 0xFFFFu) << 16);      // tile_dim0 = elems
    g1[4] = (int)(1u | (1u << 16));              // tile_dim1=1, tile_dim2=1
    g1[5] = (int)elems;                          // tensor_dim0_stride lo32
    g1[6] = (int)((elems & 0xFFFFu) << 16);      // dim0_stride hi=0 | dim1_stride lo
    g1[7] = 0;
    g2[0] = 1; g2[1] = 1; g2[2] = 0; g2[3] = (int)(1u << 16); // dims2/3=1
    g3[0] = 0; g3[1] = (int)(1u << 16); g3[2] = (int)(1u << 16); g3[3] = 0;
    g4[0] = 0; g4[1] = 0; g4[2] = 0; g4[3] = 0;
    g4[4] = 0; g4[5] = 0; g4[6] = 0; g4[7] = 0;  // VADDR4: unused
    __builtin_amdgcn_tensor_load_to_lds(g0, g1, g2, g3, g4, 0);
}
#endif

// ---------------------------------------------------------------------------
// fp32 -> f16 convert (optionally scaled): weights and the pre-scaled query.
// ---------------------------------------------------------------------------
__global__ void cvt_f32_f16_kernel(const float* __restrict__ src,
                                   half_t* __restrict__ dst,
                                   float scale, int n) {
    int i = (int)(blockIdx.x * blockDim.x + threadIdx.x);
    if (i < n) dst[i] = (half_t)(src[i] * scale);
}

// ---------------------------------------------------------------------------
// LayerNorm1: read x in original (B, H*W, C) order, write normalized x in
// WINDOWED token order, as f32 (shortcut) and f16 (GEMM operand).
// One wave per token; 8 waves / block.
// ---------------------------------------------------------------------------
__global__ void ln1_kernel(const float* __restrict__ x,
                           const float* __restrict__ w,
                           const float* __restrict__ b,
                           float* __restrict__ xn_f,
                           half_t* __restrict__ xn_h) {
    const int wave = (int)(threadIdx.x >> 5);
    const int lane = (int)(threadIdx.x & 31u);
    const int T = (int)blockIdx.x * 8 + wave;      // windowed token id
    const int b_  = T >> 7;                        // window
    const int n   = T & 127;                       // token in window
    const int bb  = b_ >> 3;                       // image
    const int win = b_ & 7;                        // window column block
    const int mh  = n >> 4;                        // row in window
    const int mw  = n & 15;                        // col in window
    const size_t src_off = ((size_t)bb * 1024 + (size_t)mh * 128 + win * 16 + mw) * CDIM;
    const float* src = x + src_off;

    float v[8];
    const float4* s4 = (const float4*)(src + lane * 8);
    float4 p0 = s4[0], p1 = s4[1];
    v[0]=p0.x; v[1]=p0.y; v[2]=p0.z; v[3]=p0.w;
    v[4]=p1.x; v[5]=p1.y; v[6]=p1.z; v[7]=p1.w;

    float s = 0.f;
#pragma unroll
    for (int j = 0; j < 8; ++j) s += v[j];
#pragma unroll
    for (int m = 1; m < 32; m <<= 1) s += __shfl_xor(s, m, 32);
    const float mu = s * (1.f / 256.f);

    float s2 = 0.f;
#pragma unroll
    for (int j = 0; j < 8; ++j) { float d = v[j] - mu; s2 += d * d; }
#pragma unroll
    for (int m = 1; m < 32; m <<= 1) s2 += __shfl_xor(s2, m, 32);
    const float rs = rsqrtf(s2 * (1.f / 256.f) + 1e-5f);

    float* of = xn_f + (size_t)T * CDIM;
    half_t* oh = xn_h + (size_t)T * CDIM;
#pragma unroll
    for (int j = 0; j < 8; ++j) {
        int c = lane * 8 + j;
        float y = (v[j] - mu) * rs * w[c] + b[c];
        of[c] = y;
        oh[c] = (half_t)y;
    }
}

// ---------------------------------------------------------------------------
// LayerNorm2: x2 (already windowed, f32) -> f16 normalized.
// ---------------------------------------------------------------------------
__global__ void ln2_kernel(const float* __restrict__ x2,
                           const float* __restrict__ w,
                           const float* __restrict__ b,
                           half_t* __restrict__ xm_h) {
    const int wave = (int)(threadIdx.x >> 5);
    const int lane = (int)(threadIdx.x & 31u);
    const int T = (int)blockIdx.x * 8 + wave;
    const float* src = x2 + (size_t)T * CDIM;

    float v[8];
    const float4* s4 = (const float4*)(src + lane * 8);
    float4 p0 = s4[0], p1 = s4[1];
    v[0]=p0.x; v[1]=p0.y; v[2]=p0.z; v[3]=p0.w;
    v[4]=p1.x; v[5]=p1.y; v[6]=p1.z; v[7]=p1.w;

    float s = 0.f;
#pragma unroll
    for (int j = 0; j < 8; ++j) s += v[j];
#pragma unroll
    for (int m = 1; m < 32; m <<= 1) s += __shfl_xor(s, m, 32);
    const float mu = s * (1.f / 256.f);

    float s2 = 0.f;
#pragma unroll
    for (int j = 0; j < 8; ++j) { float d = v[j] - mu; s2 += d * d; }
#pragma unroll
    for (int m = 1; m < 32; m <<= 1) s2 += __shfl_xor(s2, m, 32);
    const float rs = rsqrtf(s2 * (1.f / 256.f) + 1e-5f);

    half_t* oh = xm_h + (size_t)T * CDIM;
#pragma unroll
    for (int j = 0; j < 8; ++j) {
        int c = lane * 8 + j;
        oh[c] = (half_t)((v[j] - mu) * rs * w[c] + b[c]);
    }
}

// ---------------------------------------------------------------------------
// kv projection: xn (tokens x 256) @ qkv_w^T (256->512) + qkv_b.
// Scatter: cols 0..255 -> k[b_][h][n][d] (row-major keys),
//          cols 256..511 -> v^T[b_][h][d][n] (transposed for P@V B-frags).
// ---------------------------------------------------------------------------
__global__ void gemm_kv_kernel(const half_t* __restrict__ A,
                               const half_t* __restrict__ Wt,
                               const float* __restrict__ bias,
                               half_t* __restrict__ k_h,
                               half_t* __restrict__ vT_h) {
    const int wave = (int)(threadIdx.x >> 5);
    const int lane = (int)(threadIdx.x & 31u);
    const int wr = wave >> 1, wc = wave & 1;
    const int m0 = (int)blockIdx.x * 128 + wr * 32;
    const int n0 = (int)blockIdx.y * 64 + wc * 32;

    v8f acc[2][2];
#pragma unroll
    for (int i = 0; i < 2; ++i)
#pragma unroll
        for (int j = 0; j < 2; ++j) acc[i][j] = VZERO8;

    wave_gemm32x32(A + (size_t)m0 * CDIM, CDIM, Wt + (size_t)n0 * CDIM, CDIM,
                   CDIM, acc);

    const int hf = lane >> 4, cl = lane & 15;
#pragma unroll
    for (int mt = 0; mt < 2; ++mt)
#pragma unroll
        for (int nt = 0; nt < 2; ++nt)
#pragma unroll
            for (int r = 0; r < 8; ++r) {
                int gm = m0 + 16 * mt + r + 8 * hf;     // windowed token
                int gn = n0 + 16 * nt + cl;             // output col 0..511
                float val = acc[mt][nt][r] + bias[gn];
                int b_ = gm >> 7, n = gm & 127;
                if (gn < 256) {
                    int h = gn >> 5, d = gn & 31;
                    k_h[(((size_t)b_ * NHEAD + h) * NWIN + n) * HDIM + d] = (half_t)val;
                } else {
                    int j = gn - 256;
                    int h = j >> 5, d = j & 31;
                    vT_h[(((size_t)b_ * NHEAD + h) * HDIM + d) * NWIN + n] = (half_t)val;
                }
            }
}

// ---------------------------------------------------------------------------
// Windowed attention: one block per (window, head). 8 waves x 16 query rows.
// K/V tiles (8 KB each, contiguous) staged to LDS by the Tensor Data Mover;
// S = q_scaled @ k^T (+rel-pos bias) -> softmax -> O = P @ v.
// ---------------------------------------------------------------------------
__global__ void attn_kernel(const half_t* __restrict__ q_h,   // (B,nh,N,hd) pre-scaled
                            const half_t* __restrict__ k_h,   // (B_,nh,N,hd)
                            const half_t* __restrict__ vT_h,  // (B_,nh,hd,N)
                            const float* __restrict__ bias_table, // (945,nh)
                            half_t* __restrict__ attn_h) {    // (B_,N,C)
    // [0,4096) K tile, [4096,8192) V^T tile, [8192,24576) per-wave P tiles
    __shared__ _Float16 smem[24576];   // 48 KB

    const int bh = (int)blockIdx.x;
    const int b_ = bh >> 3;
    const int h  = bh & 7;
    const int bb = b_ >> 3;                   // image (q broadcast over windows)
    const int tid  = (int)threadIdx.x;
    const int wave = tid >> 5;
    const int lane = tid & 31;
    const int hf = lane >> 4, cl = lane & 15;
    const int m0 = wave * 16;

    const half_t* qb = q_h + (((size_t)bb * NHEAD + h) * NWIN + m0) * HDIM;
    const half_t* kb = k_h + (((size_t)b_ * NHEAD + h) * NWIN) * HDIM;
    const half_t* vt = vT_h + (((size_t)b_ * NHEAD + h) * HDIM) * NWIN;

    half_t* Ksm = smem;                 // 128 x 32 keys, row-major
    half_t* Vsm = smem + 4096;          // 32 x 128 v^T, row-major
    half_t* Psm = smem + 8192 + wave * (16 * NWIN);

#ifdef HAVE_TDM
    if (wave == 0) {
        // LDS byte address = low 32 bits of the generic pointer (flat LDS
        // aperture maps addr[31:0] to the LDS offset).
        tdm_load_1d((uint64_t)(uintptr_t)kb, (unsigned)(uintptr_t)Ksm, 8192u);
        tdm_load_1d((uint64_t)(uintptr_t)vt, (unsigned)(uintptr_t)Vsm, 8192u);
        __builtin_amdgcn_s_wait_tensorcnt(0);
    }
#else
    {   // cooperative fallback copy (no TDM builtin on this toolchain)
        const uint32_t* gk = (const uint32_t*)kb;
        const uint32_t* gv = (const uint32_t*)vt;
        uint32_t* lk = (uint32_t*)Ksm;
        uint32_t* lv = (uint32_t*)Vsm;
        for (int i = tid; i < 2048; i += 256) { lk[i] = gk[i]; lv[i] = gv[i]; }
    }
#endif
    __syncthreads();

    // ---- S = q @ k^T : 8 tiles of 16x16, single K=32 WMMA each ----
    v16h qa = load_a16(qb, HDIM);
    v8f accS[8];
#pragma unroll
    for (int t = 0; t < 8; ++t) {
        v16h kbf = load_b16(Ksm + (16 * t) * HDIM, HDIM);
        accS[t] = wmma_f16(qa, kbf, VZERO8);
    }

    // ---- add relative-position bias ----
#pragma unroll
    for (int t = 0; t < 8; ++t)
#pragma unroll
        for (int r = 0; r < 8; ++r) {
            int m = m0 + r + 8 * hf;
            int n = 16 * t + cl;
            int mh = m >> 4, mw = m & 15;
            int nh2 = n >> 4, nw = n & 15;
            int rpi = (mh - nh2 + 7) * BIAS_LD + (mw - nw + 31);
            accS[t][r] += bias_table[rpi * NHEAD + h];
        }

    // ---- softmax over the 128 columns of each row; row = r + 8*hf ----
#pragma unroll
    for (int r = 0; r < 8; ++r) {
        float mx = -3.0e38f;
#pragma unroll
        for (int t = 0; t < 8; ++t) mx = fmaxf(mx, accS[t][r]);
#pragma unroll
        for (int m = 1; m < 16; m <<= 1) mx = fmaxf(mx, __shfl_xor(mx, m, 32));
        float sm = 0.f;
#pragma unroll
        for (int t = 0; t < 8; ++t) {
            float e = expf(accS[t][r] - mx);
            accS[t][r] = e;
            sm += e;
        }
#pragma unroll
        for (int m = 1; m < 16; m <<= 1) sm += __shfl_xor(sm, m, 32);
        float inv = 1.f / sm;
#pragma unroll
        for (int t = 0; t < 8; ++t)
            Psm[(r + 8 * hf) * NWIN + 16 * t + cl] = (half_t)(accS[t][r] * inv);
    }
    __syncthreads();

    // ---- O = P @ v : 2 tiles (hd=32), K=128 in 4 chunks ----
    v8f oacc[2];
    oacc[0] = VZERO8; oacc[1] = VZERO8;
#pragma unroll
    for (int c = 0; c < 4; ++c) {
        v16h pa = load_a16(Psm + c * 32, NWIN);
#pragma unroll
        for (int dt = 0; dt < 2; ++dt) {
            v16h vb = load_b16(Vsm + (dt * 16) * NWIN + c * 32, NWIN);
            oacc[dt] = wmma_f16(pa, vb, oacc[dt]);
        }
    }

    // ---- store (B_, N, nh*hd) ----
#pragma unroll
    for (int dt = 0; dt < 2; ++dt)
#pragma unroll
        for (int r = 0; r < 8; ++r) {
            int row = m0 + r + 8 * hf;
            int d = dt * 16 + cl;
            attn_h[((size_t)b_ * NWIN + row) * CDIM + h * HDIM + d] =
                (half_t)oacc[dt][r];
        }
}

// ---------------------------------------------------------------------------
// proj: attn @ proj_w^T + proj_b; x2 = shortcut(xn_f) + proj -> x2_f.
// ---------------------------------------------------------------------------
__global__ void gemm_proj_kernel(const half_t* __restrict__ A,
                                 const half_t* __restrict__ Wt,
                                 const float* __restrict__ bias,
                                 const float* __restrict__ shortcut,
                                 float* __restrict__ x2_f) {
    const int wave = (int)(threadIdx.x >> 5);
    const int lane = (int)(threadIdx.x & 31u);
    const int wr = wave >> 1, wc = wave & 1;
    const int m0 = (int)blockIdx.x * 128 + wr * 32;
    const int n0 = (int)blockIdx.y * 64 + wc * 32;

    v8f acc[2][2];
#pragma unroll
    for (int i = 0; i < 2; ++i)
#pragma unroll
        for (int j = 0; j < 2; ++j) acc[i][j] = VZERO8;

    wave_gemm32x32(A + (size_t)m0 * CDIM, CDIM, Wt + (size_t)n0 * CDIM, CDIM,
                   CDIM, acc);

    const int hf = lane >> 4, cl = lane & 15;
#pragma unroll
    for (int mt = 0; mt < 2; ++mt)
#pragma unroll
        for (int nt = 0; nt < 2; ++nt)
#pragma unroll
            for (int r = 0; r < 8; ++r) {
                int gm = m0 + 16 * mt + r + 8 * hf;
                int gn = n0 + 16 * nt + cl;
                size_t off = (size_t)gm * CDIM + gn;
                x2_f[off] = shortcut[off] + acc[mt][nt][r] + bias[gn];
            }
}

// ---------------------------------------------------------------------------
// fc1: xm @ fc1_w^T + fc1_b, exact GELU -> h_h (f16).
// ---------------------------------------------------------------------------
__global__ void gemm_fc1_kernel(const half_t* __restrict__ A,
                                const half_t* __restrict__ Wt,
                                const float* __restrict__ bias,
                                half_t* __restrict__ h_h) {
    const int wave = (int)(threadIdx.x >> 5);
    const int lane = (int)(threadIdx.x & 31u);
    const int wr = wave >> 1, wc = wave & 1;
    const int m0 = (int)blockIdx.x * 128 + wr * 32;
    const int n0 = (int)blockIdx.y * 64 + wc * 32;

    v8f acc[2][2];
#pragma unroll
    for (int i = 0; i < 2; ++i)
#pragma unroll
        for (int j = 0; j < 2; ++j) acc[i][j] = VZERO8;

    wave_gemm32x32(A + (size_t)m0 * CDIM, CDIM, Wt + (size_t)n0 * CDIM, CDIM,
                   CDIM, acc);

    const int hf = lane >> 4, cl = lane & 15;
#pragma unroll
    for (int mt = 0; mt < 2; ++mt)
#pragma unroll
        for (int nt = 0; nt < 2; ++nt)
#pragma unroll
            for (int r = 0; r < 8; ++r) {
                int gm = m0 + 16 * mt + r + 8 * hf;
                int gn = n0 + 16 * nt + cl;
                float v = acc[mt][nt][r] + bias[gn];
                float g = 0.5f * v * (1.f + erff(v * 0.70710678118654752f));
                h_h[(size_t)gm * DFF + gn] = (half_t)g;
            }
}

// ---------------------------------------------------------------------------
// fc2: h @ fc2_w^T + fc2_b + x2 -> final, window-reversed into d_out (f32).
// ---------------------------------------------------------------------------
__global__ void gemm_fc2_kernel(const half_t* __restrict__ A,
                                const half_t* __restrict__ Wt,
                                const float* __restrict__ bias,
                                const float* __restrict__ x2_f,
                                float* __restrict__ out) {
    const int wave = (int)(threadIdx.x >> 5);
    const int lane = (int)(threadIdx.x & 31u);
    const int wr = wave >> 1, wc = wave & 1;
    const int m0 = (int)blockIdx.x * 128 + wr * 32;
    const int n0 = (int)blockIdx.y * 64 + wc * 32;

    v8f acc[2][2];
#pragma unroll
    for (int i = 0; i < 2; ++i)
#pragma unroll
        for (int j = 0; j < 2; ++j) acc[i][j] = VZERO8;

    wave_gemm32x32(A + (size_t)m0 * DFF, DFF, Wt + (size_t)n0 * DFF, DFF,
                   DFF, acc);

    const int hf = lane >> 4, cl = lane & 15;
#pragma unroll
    for (int mt = 0; mt < 2; ++mt)
#pragma unroll
        for (int nt = 0; nt < 2; ++nt)
#pragma unroll
            for (int r = 0; r < 8; ++r) {
                int gm = m0 + 16 * mt + r + 8 * hf;   // windowed token
                int gn = n0 + 16 * nt + cl;
                float v = acc[mt][nt][r] + bias[gn] + x2_f[(size_t)gm * CDIM + gn];
                // window-reverse: T -> (b, mh*W + win*16 + mw)
                int b_ = gm >> 7, n = gm & 127;
                int bb = b_ >> 3, win = b_ & 7;
                int mh = n >> 4, mw = n & 15;
                size_t pos = (size_t)bb * 1024 + (size_t)mh * 128 + win * 16 + mw;
                out[pos * CDIM + gn] = v;
            }
}

// ---------------------------------------------------------------------------
// Host launcher
// ---------------------------------------------------------------------------
extern "C" void kernel_launch(void* const* d_in, const int* in_sizes, int n_in,
                              void* d_out, int out_size, void* d_ws, size_t ws_size,
                              hipStream_t stream) {
    const float* x          = (const float*)d_in[0];
    const float* q_global   = (const float*)d_in[1];
    const float* norm1_w    = (const float*)d_in[2];
    const float* norm1_b    = (const float*)d_in[3];
    const float* qkv_w      = (const float*)d_in[4];
    const float* qkv_b      = (const float*)d_in[5];
    const float* bias_table = (const float*)d_in[6];
    const float* proj_w     = (const float*)d_in[7];
    const float* proj_b     = (const float*)d_in[8];
    const float* norm2_w    = (const float*)d_in[9];
    const float* norm2_b    = (const float*)d_in[10];
    const float* fc1_w      = (const float*)d_in[11];
    const float* fc1_b      = (const float*)d_in[12];
    const float* fc2_w      = (const float*)d_in[13];
    const float* fc2_b      = (const float*)d_in[14];
    float* out = (float*)d_out;

    // workspace carve-out (256B aligned)
    size_t off = 0;
    auto carve = [&](size_t bytes) -> void* {
        void* p = (char*)d_ws + off;
        off += (bytes + 255) & ~(size_t)255;
        return p;
    };
    half_t* xn_h   = (half_t*)carve((size_t)TOKENS * CDIM * 2);
    float*  xn_f   = (float*) carve((size_t)TOKENS * CDIM * 4);
    half_t* wqkv_h = (half_t*)carve((size_t)512 * 256 * 2);
    half_t* wproj_h= (half_t*)carve((size_t)256 * 256 * 2);
    half_t* wfc1_h = (half_t*)carve((size_t)1024 * 256 * 2);
    half_t* wfc2_h = (half_t*)carve((size_t)256 * 1024 * 2);
    half_t* q_h    = (half_t*)carve((size_t)64 * NHEAD * NWIN * HDIM * 2);
    half_t* k_h    = (half_t*)carve((size_t)BWIN * NHEAD * NWIN * HDIM * 2);
    half_t* vT_h   = (half_t*)carve((size_t)BWIN * NHEAD * HDIM * NWIN * 2);
    half_t* attn_h = (half_t*)carve((size_t)TOKENS * CDIM * 2);
    float*  x2_f   = (float*) carve((size_t)TOKENS * CDIM * 4);
    half_t* xm_h   = (half_t*)carve((size_t)TOKENS * CDIM * 2);
    half_t* h_h    = (half_t*)carve((size_t)TOKENS * DFF * 2);

    const float qscale = 0.17677669529663688f; // 32^-0.5

    // weight / query conversion to f16
    cvt_f32_f16_kernel<<<(512*256 + 255)/256, 256, 0, stream>>>(qkv_w, wqkv_h, 1.f, 512*256);
    cvt_f32_f16_kernel<<<(256*256 + 255)/256, 256, 0, stream>>>(proj_w, wproj_h, 1.f, 256*256);
    cvt_f32_f16_kernel<<<(1024*256 + 255)/256, 256, 0, stream>>>(fc1_w, wfc1_h, 1.f, 1024*256);
    cvt_f32_f16_kernel<<<(256*1024 + 255)/256, 256, 0, stream>>>(fc2_w, wfc2_h, 1.f, 256*1024);
    cvt_f32_f16_kernel<<<(64*NHEAD*NWIN*HDIM + 255)/256, 256, 0, stream>>>(
        q_global, q_h, qscale, 64*NHEAD*NWIN*HDIM);

    // LN1 (windowed layout out)
    ln1_kernel<<<TOKENS/8, 256, 0, stream>>>(x, norm1_w, norm1_b, xn_f, xn_h);

    // kv projection
    gemm_kv_kernel<<<dim3(TOKENS/128, 512/64), 256, 0, stream>>>(
        xn_h, wqkv_h, qkv_b, k_h, vT_h);

    // windowed attention (TDM-staged K/V)
    attn_kernel<<<BWIN * NHEAD, 256, 0, stream>>>(
        q_h, k_h, vT_h, bias_table, attn_h);

    // proj + residual
    gemm_proj_kernel<<<dim3(TOKENS/128, CDIM/64), 256, 0, stream>>>(
        attn_h, wproj_h, proj_b, xn_f, x2_f);

    // LN2
    ln2_kernel<<<TOKENS/8, 256, 0, stream>>>(x2_f, norm2_w, norm2_b, xm_h);

    // MLP
    gemm_fc1_kernel<<<dim3(TOKENS/128, DFF/64), 256, 0, stream>>>(
        xm_h, wfc1_h, fc1_b, h_h);
    gemm_fc2_kernel<<<dim3(TOKENS/128, CDIM/64), 256, 0, stream>>>(
        h_h, wfc2_h, fc2_b, x2_f, out);

    (void)in_sizes; (void)n_in; (void)out_size; (void)ws_size;
}
